// GraphDecoder_72859825209531
// MI455X (gfx1250) — matrix operand order repaired and checked by reading, not compiled, not run on hardware
//
#include <hip/hip_runtime.h>
#include <hip/hip_bf16.h>

typedef __bf16 bf16;
typedef __bf16 v16bf __attribute__((ext_vector_type(16)));
typedef float  v8f   __attribute__((ext_vector_type(8)));
typedef unsigned int u32x4 __attribute__((ext_vector_type(4)));

#define BATCH  16384
#define NSTEPS 50
#define BM     32
#define HST    264   // padded LDS row stride (bf16 elems) to avoid bank conflicts

union Frag16 { v16bf v; u32x4 q[2]; };
union F8     { v8f   v; u32x4 q[2]; };

__device__ __forceinline__ v8f wmma_bf16(v16bf a, v16bf b, v8f c) {
  return __builtin_amdgcn_wmma_f32_16x16x32_bf16(false, a, false, b, (short)0, c, false, false);
}

// B fragment (32x16 K x N, bf16), pre-packed: lane's 16 bf16 contiguous (32B)
__device__ __forceinline__ v16bf load_bfrag(const bf16* frags, int fidx, int lane) {
  Frag16 f;
  const u32x4* p = (const u32x4*)(frags + (size_t)fidx * 512) + lane * 2;
  f.q[0] = p[0]; f.q[1] = p[1];
  return f.v;
}

// A fragment (16x32 M x K, bf16) from row-major [rows, ldm] storage.
// lanes 0-15: M=lane, K=kt*32+{0..15}; lanes 16-31: M=lane-16, K=+16
__device__ __forceinline__ v16bf load_afrag(const bf16* base, int ldm, int row0, int kt, int lane) {
  Frag16 f;
  int m  = row0 + (lane & 15);
  int kb = kt * 32 + (lane & 16);              // 0 or 16
  const u32x4* p = (const u32x4*)(base + (size_t)m * ldm + kb);
  f.q[0] = p[0];   // K = kb .. kb+7
  f.q[1] = p[1];   // K = kb+8 .. kb+15
  return f.v;
}

__device__ __forceinline__ float sigm(float x) { return 1.0f / (1.0f + __expf(-x)); }
__device__ __forceinline__ float tanh_f(float x) {
  x = fminf(fmaxf(x, -15.f), 15.f);
  float e = __expf(2.f * x);
  return (e - 1.f) / (e + 1.f);
}
__device__ __forceinline__ v8f splat8(float s) {
  v8f r;
  #pragma unroll
  for (int i = 0; i < 8; ++i) r[i] = s;
  return r;
}

// ---------------- pack kernels ----------------

__global__ void k_pack_zc(const float* __restrict__ z, const float* __restrict__ cond,
                          bf16* __restrict__ zcb, int n) {
  int i = blockIdx.x * blockDim.x + threadIdx.x;
  if (i >= n) return;
  int row = i / 288, c = i - row * 288;
  float v = 0.f;
  if (c < 256)      v = z[(size_t)row * 256 + c];
  else if (c < 260) v = cond[(size_t)row * 4 + (c - 256)];
  zcb[i] = (bf16)v;
}

__global__ void k_pack_prev(const float* __restrict__ tgt, bf16* __restrict__ prevb, int n) {
  int i = blockIdx.x * blockDim.x + threadIdx.x;
  if (i >= n) return;
  int c = i & 31;
  int rt = i >> 5;
  int row = rt % BATCH;
  int t = rt / BATCH;
  float v = 0.f;
  if (t > 0 && c < 29) v = tgt[((size_t)row * NSTEPS + (t - 1)) * 29 + c];
  prevb[i] = (bf16)v;
}

__global__ void k_pack_b(const float* __restrict__ src, bf16* __restrict__ dst,
                         int Ksrc, int Nsrc, int ldn, int KT, int NT) {
  int i = blockIdx.x * blockDim.x + threadIdx.x;
  int total = NT * KT * 512;
  if (i >= total) return;
  int f = i >> 9, q = i & 511, lane = q >> 4, e = q & 15;
  int kt = f % KT, nt = f / KT;
  int K = kt * 32 + ((lane & 16) ? 16 : 0) + e;
  int N = nt * 16 + (lane & 15);
  float v = (K < Ksrc && N < Nsrc) ? src[(size_t)K * ldn + N] : 0.f;
  dst[i] = (bf16)v;
}

__global__ void k_pack_we1sum(const float* __restrict__ We1, bf16* __restrict__ dst) {
  // pair = [h, h, zc] -> first two 256-row blocks of We1 collapse into a sum
  int i = blockIdx.x * blockDim.x + threadIdx.x;
  int total = 16 * 8 * 512;                 // NT=16, KT=8
  if (i >= total) return;
  int f = i >> 9, q = i & 511, lane = q >> 4, e = q & 15;
  int kt = f % 8, nt = f / 8;
  int K = kt * 32 + ((lane & 16) ? 16 : 0) + e;
  int N = nt * 16 + (lane & 15);
  float v = (K < 256 && N < 256)
              ? We1[(size_t)K * 256 + N] + We1[(size_t)(256 + K) * 256 + N] : 0.f;
  dst[i] = (bf16)v;
}

__global__ void k_pack_wtp(const float* __restrict__ Wt, const float* __restrict__ Wp,
                           bf16* __restrict__ dst) {
  int i = blockIdx.x * blockDim.x + threadIdx.x;
  int total = 2 * 8 * 512;                  // NT=2, KT=8
  if (i >= total) return;
  int f = i >> 9, q = i & 511, lane = q >> 4, e = q & 15;
  int kt = f % 8, nt = f / 8;
  int K = kt * 32 + ((lane & 16) ? 16 : 0) + e;
  int N = nt * 16 + (lane & 15);
  float v = 0.f;
  if (N < 16) { if (N < 15) v = Wt[(size_t)K * 15 + N]; }
  else        { int c = N - 16; if (c < 14) v = Wp[(size_t)K * 14 + c]; }
  dst[i] = (bf16)v;
}

__global__ void k_pack_bias(const float* __restrict__ bih, const float* __restrict__ bhh,
                            float* __restrict__ brz, float* __restrict__ bni,
                            float* __restrict__ bnh) {
  int i = blockIdx.x * blockDim.x + threadIdx.x;
  if (i < 512) brz[i] = bih[i] + bhh[i];
  if (i < 256) { bni[i] = bih[512 + i]; bnh[i] = bhh[512 + i]; }
}

// ---------------- main recurrent kernel ----------------
// 1 block = 32 batch rows, 8 waves; wave w owns hidden columns [w*32, w*32+32)

__global__ __launch_bounds__(256) void k_decoder(
    const bf16* __restrict__ zcb, const bf16* __restrict__ prevb,
    const bf16* __restrict__ Whh_p, const bf16* __restrict__ Wzc_p, const bf16* __restrict__ Wpr_p,
    const bf16* __restrict__ Wi_p, const bf16* __restrict__ Wn1_p, const bf16* __restrict__ Wn2_p,
    const bf16* __restrict__ Wtp_p, const bf16* __restrict__ We1s_p, const bf16* __restrict__ We1c_p,
    const float* __restrict__ brz, const float* __restrict__ bni, const float* __restrict__ bnh,
    const float* __restrict__ bi, const float* __restrict__ bn1, const float* __restrict__ bn2,
    const float* __restrict__ bt, const float* __restrict__ bp, const float* __restrict__ be1,
    const float* __restrict__ We2, const float* __restrict__ be2,
    float* __restrict__ out_nn, float* __restrict__ out_ty,
    float* __restrict__ out_pa, float* __restrict__ out_ed)
{
  __shared__ alignas(16) bf16  h_lds[BM * HST];        // ~16.9 KB (h_t as bf16)
  __shared__ alignas(16) float gz_lds[8 * 12 * 256];   // 96 KB (zc@Wih + biases, frag layout)
  __shared__ alignas(16) float scratch[BM * HST];      // 33.8 KB (t1 bf16 / e1 f32, reused)
  __shared__ float e_sm[BM];
  bf16* t1_lds = (bf16*)scratch;

  const int tid  = threadIdx.x;
  const int lane = tid & 31;
  const int w    = tid >> 5;
  const int ncol = lane & 15;
  const int hi8  = (lane & 16) >> 1;      // C-layout: upper lanes hold rows +8
  const int rowBase = blockIdx.x * BM;
  const int colW = w * 32;

  const float bnh_l0 = bnh[colW + ncol];
  const float bnh_l1 = bnh[colW + 16 + ncol];
  float btp = 0.f;
  if (w == 0 && ncol < 15) btp = bt[ncol];
  if (w == 1 && ncol < 14) btp = bp[ncol];

  // per-wave fragment strides: gate-matrix fragment id = (reg*16 + w*2 + j)*KT + kt
  // rj = reg*2 + j  (6 column tiles per wave across r/z/n regions)

  // ---- gz = zc @ Wih[0:260] + biases (12 fragments per wave: r,z combined + n_input) ----
  {
    v8f acc[12];
    #pragma unroll
    for (int reg = 0; reg < 3; ++reg)
      #pragma unroll
      for (int j = 0; j < 2; ++j) {
        float b = (reg < 2) ? brz[reg * 256 + colW + j * 16 + ncol]
                            : bni[colW + j * 16 + ncol];
        acc[reg * 4 + j * 2 + 0] = splat8(b);
        acc[reg * 4 + j * 2 + 1] = splat8(b);
      }
    for (int kt = 0; kt < 9; ++kt) {
      v16bf bfr[6];
      #pragma unroll
      for (int rj = 0; rj < 6; ++rj)
        bfr[rj] = load_bfrag(Wzc_p, ((rj >> 1) * 16 + colW / 16 + (rj & 1)) * 9 + kt, lane);
      v16bf a0 = load_afrag(zcb, 288, rowBase, kt, lane);
      v16bf a1 = load_afrag(zcb, 288, rowBase + 16, kt, lane);
      #pragma unroll
      for (int rj = 0; rj < 6; ++rj) {
        acc[(rj >> 1) * 4 + (rj & 1) * 2 + 0] = wmma_bf16(a0, bfr[rj], acc[(rj >> 1) * 4 + (rj & 1) * 2 + 0]);
        acc[(rj >> 1) * 4 + (rj & 1) * 2 + 1] = wmma_bf16(a1, bfr[rj], acc[(rj >> 1) * 4 + (rj & 1) * 2 + 1]);
      }
    }
    u32x4* gp = (u32x4*)(gz_lds + (size_t)(w * 12) * 256) + lane * 2;
    #pragma unroll
    for (int f = 0; f < 12; ++f) {
      F8 x; x.v = acc[f];
      gp[f * 64 + 0] = x.q[0];
      gp[f * 64 + 1] = x.q[1];
    }
  }

  // ---- h0 = zc @ Wi + bi ; kept in registers (f32) and LDS (bf16) ----
  v8f hreg[2][2];
  {
    v8f acc[4];
    #pragma unroll
    for (int j = 0; j < 2; ++j) {
      float b = bi[colW + j * 16 + ncol];
      acc[j * 2 + 0] = splat8(b);
      acc[j * 2 + 1] = splat8(b);
    }
    for (int kt = 0; kt < 9; ++kt) {
      v16bf bfr[2];
      #pragma unroll
      for (int j = 0; j < 2; ++j)
        bfr[j] = load_bfrag(Wi_p, (w * 2 + j) * 9 + kt, lane);
      v16bf a0 = load_afrag(zcb, 288, rowBase, kt, lane);
      v16bf a1 = load_afrag(zcb, 288, rowBase + 16, kt, lane);
      #pragma unroll
      for (int j = 0; j < 2; ++j) {
        acc[j * 2 + 0] = wmma_bf16(a0, bfr[j], acc[j * 2 + 0]);
        acc[j * 2 + 1] = wmma_bf16(a1, bfr[j], acc[j * 2 + 1]);
      }
    }
    #pragma unroll
    for (int j = 0; j < 2; ++j)
      #pragma unroll
      for (int mt = 0; mt < 2; ++mt) {
        hreg[j][mt] = acc[j * 2 + mt];
        #pragma unroll
        for (int v = 0; v < 8; ++v)
          h_lds[(mt * 16 + v + hi8) * HST + colW + j * 16 + ncol] = (bf16)acc[j * 2 + mt][v];
      }
  }

  // ---- num_nodes head: t1 = relu(zc@Wn1 + bn1); logits = t1@Wn2 + bn2 ----
  {
    v8f acc[4];
    #pragma unroll
    for (int j = 0; j < 2; ++j) {
      float b = bn1[colW + j * 16 + ncol];
      acc[j * 2 + 0] = splat8(b);
      acc[j * 2 + 1] = splat8(b);
    }
    for (int kt = 0; kt < 9; ++kt) {
      v16bf bfr[2];
      #pragma unroll
      for (int j = 0; j < 2; ++j)
        bfr[j] = load_bfrag(Wn1_p, (w * 2 + j) * 9 + kt, lane);
      v16bf a0 = load_afrag(zcb, 288, rowBase, kt, lane);
      v16bf a1 = load_afrag(zcb, 288, rowBase + 16, kt, lane);
      #pragma unroll
      for (int j = 0; j < 2; ++j) {
        acc[j * 2 + 0] = wmma_bf16(a0, bfr[j], acc[j * 2 + 0]);
        acc[j * 2 + 1] = wmma_bf16(a1, bfr[j], acc[j * 2 + 1]);
      }
    }
    #pragma unroll
    for (int j = 0; j < 2; ++j)
      #pragma unroll
      for (int mt = 0; mt < 2; ++mt)
        #pragma unroll
        for (int v = 0; v < 8; ++v)
          t1_lds[(mt * 16 + v + hi8) * HST + colW + j * 16 + ncol] =
              (bf16)fmaxf(acc[j * 2 + mt][v], 0.f);
  }
  __syncthreads();
  if (w < 4) {                                    // Wn2: [256, 49->64], waves 0..3
    int col = w * 16 + ncol;
    float bb = (col < 49) ? bn2[col] : 0.f;
    v8f acc0 = splat8(bb), acc1 = splat8(bb);
    for (int kt = 0; kt < 8; ++kt) {
      v16bf bf = load_bfrag(Wn2_p, w * 8 + kt, lane);
      v16bf a0 = load_afrag(t1_lds, HST, 0, kt, lane);
      v16bf a1 = load_afrag(t1_lds, HST, 16, kt, lane);
      acc0 = wmma_bf16(a0, bf, acc0);
      acc1 = wmma_bf16(a1, bf, acc1);
    }
    if (col < 49) {
      #pragma unroll
      for (int v = 0; v < 8; ++v) {
        out_nn[(size_t)(rowBase + v + hi8) * 49 + col]      = acc0[v];
        out_nn[(size_t)(rowBase + 16 + v + hi8) * 49 + col] = acc1[v];
      }
    }
  }
  __syncthreads();

  // ---- GRU time loop ----
  for (int t = 0; t < NSTEPS; ++t) {
    const bf16* pv = prevb + (size_t)t * BATCH * 32;

    // prefetch next step's x_prev tile (32 rows x 64B, one row per lane)
    if (t + 1 < NSTEPS)
      __builtin_prefetch(prevb + (size_t)(t + 1) * BATCH * 32 + (size_t)(rowBase + lane) * 32, 0, 3);

    v8f arz[8], ani[4], anh[4];
    {   // reload per-wave gz fragments (vector ds loads)
      const u32x4* gp = (const u32x4*)(gz_lds + (size_t)(w * 12) * 256) + lane * 2;
      #pragma unroll
      for (int f = 0; f < 8; ++f) {
        F8 x; x.q[0] = gp[f * 64 + 0]; x.q[1] = gp[f * 64 + 1];
        arz[f] = x.v;
      }
      #pragma unroll
      for (int f = 0; f < 4; ++f) {
        F8 x; x.q[0] = gp[(8 + f) * 64 + 0]; x.q[1] = gp[(8 + f) * 64 + 1];
        ani[f] = x.v;
      }
    }
    anh[0] = splat8(bnh_l0); anh[1] = splat8(bnh_l0);
    anh[2] = splat8(bnh_l1); anh[3] = splat8(bnh_l1);

    {   // x_prev contribution (K=32, one WMMA step)
      v16bf bfr[6];
      #pragma unroll
      for (int rj = 0; rj < 6; ++rj)
        bfr[rj] = load_bfrag(Wpr_p, (rj >> 1) * 16 + colW / 16 + (rj & 1), lane);
      v16bf a0 = load_afrag(pv, 32, rowBase, 0, lane);
      v16bf a1 = load_afrag(pv, 32, rowBase + 16, 0, lane);
      #pragma unroll
      for (int rj = 0; rj < 6; ++rj) {
        int reg = rj >> 1, j = rj & 1;
        if (reg < 2) {
          arz[reg * 4 + j * 2 + 0] = wmma_bf16(a0, bfr[rj], arz[reg * 4 + j * 2 + 0]);
          arz[reg * 4 + j * 2 + 1] = wmma_bf16(a1, bfr[rj], arz[reg * 4 + j * 2 + 1]);
        } else {
          ani[j * 2 + 0] = wmma_bf16(a0, bfr[rj], ani[j * 2 + 0]);
          ani[j * 2 + 1] = wmma_bf16(a1, bfr[rj], ani[j * 2 + 1]);
        }
      }
    }
    // h @ Whh (K=256 -> 8 WMMA K-steps); n-region kept separate for r-gating.
    // All 6 B fragments are loaded up-front per K-step (one clause, one wait),
    // and the next K-step's fragments are prefetched while WMMAs run.
    for (int kt = 0; kt < 8; ++kt) {
      v16bf bfr[6];
      #pragma unroll
      for (int rj = 0; rj < 6; ++rj)
        bfr[rj] = load_bfrag(Whh_p, ((rj >> 1) * 16 + colW / 16 + (rj & 1)) * 8 + kt, lane);
      v16bf a0 = load_afrag(h_lds, HST, 0, kt, lane);
      v16bf a1 = load_afrag(h_lds, HST, 16, kt, lane);
      if (kt < 7) {
        #pragma unroll
        for (int rj = 0; rj < 6; ++rj)
          __builtin_prefetch(Whh_p + (size_t)(((rj >> 1) * 16 + colW / 16 + (rj & 1)) * 8 + kt + 1) * 512
                                   + lane * 16, 0, 3);
      }
      #pragma unroll
      for (int rj = 0; rj < 6; ++rj) {
        int reg = rj >> 1, j = rj & 1;
        if (reg < 2) {
          arz[reg * 4 + j * 2 + 0] = wmma_bf16(a0, bfr[rj], arz[reg * 4 + j * 2 + 0]);
          arz[reg * 4 + j * 2 + 1] = wmma_bf16(a1, bfr[rj], arz[reg * 4 + j * 2 + 1]);
        } else {
          anh[j * 2 + 0] = wmma_bf16(a0, bfr[rj], anh[j * 2 + 0]);
          anh[j * 2 + 1] = wmma_bf16(a1, bfr[rj], anh[j * 2 + 1]);
        }
      }
    }
    __syncthreads();   // all reads of h_t done
    #pragma unroll
    for (int j = 0; j < 2; ++j)
      #pragma unroll
      for (int mt = 0; mt < 2; ++mt) {
        const int idx = j * 2 + mt;
        #pragma unroll
        for (int v = 0; v < 8; ++v) {
          float r  = sigm(arz[idx][v]);
          float zg = sigm(arz[4 + idx][v]);
          float nn = tanh_f(ani[idx][v] + r * anh[idx][v]);
          float hv = (1.f - zg) * nn + zg * hreg[j][mt][v];
          hreg[j][mt][v] = hv;
          h_lds[(mt * 16 + v + hi8) * HST + colW + j * 16 + ncol] = (bf16)hv;
        }
      }
    __syncthreads();   // h_{t+1} visible

    // type/param heads on h_new: wave 0 -> types, wave 1 -> params
    if (w < 2) {
      v8f hd0 = splat8(btp), hd1 = splat8(btp);
      #pragma unroll
      for (int kt = 0; kt < 8; ++kt) {
        v16bf bf = load_bfrag(Wtp_p, w * 8 + kt, lane);
        v16bf a0 = load_afrag(h_lds, HST, 0, kt, lane);
        v16bf a1 = load_afrag(h_lds, HST, 16, kt, lane);
        hd0 = wmma_bf16(a0, bf, hd0);
        hd1 = wmma_bf16(a1, bf, hd1);
      }
      #pragma unroll
      for (int mt = 0; mt < 2; ++mt) {
        v8f hd = mt ? hd1 : hd0;
        #pragma unroll
        for (int v = 0; v < 8; ++v) {
          int row = rowBase + mt * 16 + v + hi8;
          if (w == 0 && ncol < 15) out_ty[((size_t)row * NSTEPS + t) * 15 + ncol] = hd[v];
          if (w == 1 && ncol < 14) out_pa[((size_t)row * NSTEPS + t) * 14 + ncol] = sigm(hd[v]);
        }
      }
    }
  }

  // ---- edge head: e1 = relu(h@(We1a+We1b) + zc@We1c + be1); e = e1@We2 + be2 ----
  {
    v8f acc[4];
    #pragma unroll
    for (int j = 0; j < 2; ++j) {
      float b = be1[colW + j * 16 + ncol];
      acc[j * 2 + 0] = splat8(b);
      acc[j * 2 + 1] = splat8(b);
    }
    for (int kt = 0; kt < 8; ++kt) {
      v16bf bfr[2];
      #pragma unroll
      for (int j = 0; j < 2; ++j)
        bfr[j] = load_bfrag(We1s_p, (w * 2 + j) * 8 + kt, lane);
      v16bf a0 = load_afrag(h_lds, HST, 0, kt, lane);
      v16bf a1 = load_afrag(h_lds, HST, 16, kt, lane);
      #pragma unroll
      for (int j = 0; j < 2; ++j) {
        acc[j * 2 + 0] = wmma_bf16(a0, bfr[j], acc[j * 2 + 0]);
        acc[j * 2 + 1] = wmma_bf16(a1, bfr[j], acc[j * 2 + 1]);
      }
    }
    for (int kt = 0; kt < 9; ++kt) {
      v16bf bfr[2];
      #pragma unroll
      for (int j = 0; j < 2; ++j)
        bfr[j] = load_bfrag(We1c_p, (w * 2 + j) * 9 + kt, lane);
      v16bf a0 = load_afrag(zcb, 288, rowBase, kt, lane);
      v16bf a1 = load_afrag(zcb, 288, rowBase + 16, kt, lane);
      #pragma unroll
      for (int j = 0; j < 2; ++j) {
        acc[j * 2 + 0] = wmma_bf16(a0, bfr[j], acc[j * 2 + 0]);
        acc[j * 2 + 1] = wmma_bf16(a1, bfr[j], acc[j * 2 + 1]);
      }
    }
    __syncthreads();
    #pragma unroll
    for (int j = 0; j < 2; ++j)
      #pragma unroll
      for (int mt = 0; mt < 2; ++mt)
        #pragma unroll
        for (int v = 0; v < 8; ++v)
          scratch[(mt * 16 + v + hi8) * HST + colW + j * 16 + ncol] =
              fmaxf(acc[j * 2 + mt][v], 0.f);
    __syncthreads();
    if (tid < BM) {
      float s = be2[0];
      for (int c = 0; c < 256; ++c) s += scratch[tid * HST + c] * We2[c];
      e_sm[tid] = s;
    }
    __syncthreads();
    const int NE = NSTEPS * (NSTEPS - 1);   // 2450
    for (int i = tid; i < BM * NE; i += 256) {
      int r = i / NE;
      out_ed[(size_t)(rowBase + r) * NE + (i - r * NE)] = e_sm[r];
    }
  }
}

// ---------------- launch ----------------

extern "C" void kernel_launch(void* const* d_in, const int* in_sizes, int n_in,
                              void* d_out, int out_size, void* d_ws, size_t ws_size,
                              hipStream_t stream) {
  (void)in_sizes; (void)n_in; (void)out_size; (void)ws_size;

  const float* z    = (const float*)d_in[0];
  const float* cond = (const float*)d_in[1];
  const float* tgt  = (const float*)d_in[2];
  const float* Wn1  = (const float*)d_in[3];
  const float* bn1  = (const float*)d_in[4];
  const float* Wn2  = (const float*)d_in[5];
  const float* bn2  = (const float*)d_in[6];
  const float* Wi   = (const float*)d_in[7];
  const float* bi   = (const float*)d_in[8];
  const float* Wih  = (const float*)d_in[9];
  const float* Whh  = (const float*)d_in[10];
  const float* bih  = (const float*)d_in[11];
  const float* bhh  = (const float*)d_in[12];
  const float* Wt   = (const float*)d_in[13];
  const float* bt   = (const float*)d_in[14];
  const float* Wp   = (const float*)d_in[15];
  const float* bp   = (const float*)d_in[16];
  const float* We1  = (const float*)d_in[17];
  const float* be1  = (const float*)d_in[18];
  const float* We2  = (const float*)d_in[19];
  const float* be2  = (const float*)d_in[20];

  char* ws = (char*)d_ws;
  size_t off = 0;
  auto carve = [&](size_t bytes) -> char* {
    char* p = ws + off;
    off = (off + bytes + 1023) & ~(size_t)1023;
    return p;
  };

  bf16* zcb    = (bf16*)carve((size_t)BATCH * 288 * 2);
  bf16* prevb  = (bf16*)carve((size_t)NSTEPS * BATCH * 32 * 2);
  bf16* Whh_p  = (bf16*)carve((size_t)48 * 8 * 512 * 2);
  bf16* Wzc_p  = (bf16*)carve((size_t)48 * 9 * 512 * 2);
  bf16* Wpr_p  = (bf16*)carve((size_t)48 * 1 * 512 * 2);
  bf16* Wi_p   = (bf16*)carve((size_t)16 * 9 * 512 * 2);
  bf16* Wn1_p  = (bf16*)carve((size_t)16 * 9 * 512 * 2);
  bf16* Wn2_p  = (bf16*)carve((size_t)4 * 8 * 512 * 2);
  bf16* Wtp_p  = (bf16*)carve((size_t)2 * 8 * 512 * 2);
  bf16* We1s_p = (bf16*)carve((size_t)16 * 8 * 512 * 2);
  bf16* We1c_p = (bf16*)carve((size_t)16 * 9 * 512 * 2);
  float* brz   = (float*)carve(512 * 4);
  float* bni   = (float*)carve(256 * 4);
  float* bnhv  = (float*)carve(256 * 4);

  const int tp = 256;
  auto nb = [&](long n) { return (int)((n + tp - 1) / tp); };

  k_pack_zc<<<nb((long)BATCH * 288), tp, 0, stream>>>(z, cond, zcb, BATCH * 288);
  k_pack_prev<<<nb((long)NSTEPS * BATCH * 32), tp, 0, stream>>>(tgt, prevb, NSTEPS * BATCH * 32);
  k_pack_b<<<nb(48L * 8 * 512), tp, 0, stream>>>(Whh, Whh_p, 256, 768, 768, 8, 48);
  k_pack_b<<<nb(48L * 9 * 512), tp, 0, stream>>>(Wih, Wzc_p, 260, 768, 768, 9, 48);
  k_pack_b<<<nb(48L * 1 * 512), tp, 0, stream>>>(Wih + 260 * 768, Wpr_p, 29, 768, 768, 1, 48);
  k_pack_b<<<nb(16L * 9 * 512), tp, 0, stream>>>(Wi, Wi_p, 260, 256, 256, 9, 16);
  k_pack_b<<<nb(16L * 9 * 512), tp, 0, stream>>>(Wn1, Wn1_p, 260, 256, 256, 9, 16);
  k_pack_b<<<nb(4L * 8 * 512), tp, 0, stream>>>(Wn2, Wn2_p, 256, 49, 49, 8, 4);
  k_pack_we1sum<<<nb(16L * 8 * 512), tp, 0, stream>>>(We1, We1s_p);
  k_pack_b<<<nb(16L * 9 * 512), tp, 0, stream>>>(We1 + 512 * 256, We1c_p, 260, 256, 256, 9, 16);
  k_pack_wtp<<<nb(2L * 8 * 512), tp, 0, stream>>>(Wt, Wp, Wtp_p);
  k_pack_bias<<<2, 256, 0, stream>>>(bih, bhh, brz, bni, bnhv);

  float* out    = (float*)d_out;
  float* out_nn = out;
  float* out_ty = out + (size_t)BATCH * 49;
  float* out_pa = out_ty + (size_t)BATCH * 750;
  float* out_ed = out_pa + (size_t)BATCH * 700;

  k_decoder<<<BATCH / BM, 256, 0, stream>>>(
      zcb, prevb, Whh_p, Wzc_p, Wpr_p, Wi_p, Wn1_p, Wn2_p, Wtp_p, We1s_p, We1c_p,
      brz, bni, bnhv, bi, bn1, bn2, bt, bp, be1, We2, be2,
      out_nn, out_ty, out_pa, out_ed);
}
